// Generator_83442624627354
// MI455X (gfx1250) — matrix operand order repaired and checked by reading, not compiled, not run on hardware
//
#include <hip/hip_runtime.h>
#include <hip/hip_bf16.h>
#include <cmath>

typedef __attribute__((ext_vector_type(16))) _Float16 v16h;
typedef __attribute__((ext_vector_type(8)))  _Float16 v8h;
typedef __attribute__((ext_vector_type(8)))  float    v8f;

#define LRELU 0.2f
#define NB 16          // batch
#define WDIM 256

__device__ __forceinline__ float leaky_f(float v) { return v >= 0.0f ? v : LRELU * v; }

// ---------------- style linear: s[b,i] = (w[b,:] . ts_w[i,:]) / sqrt(256) + ts_b[i] ----------
__global__ void style_kernel(const float* __restrict__ wlat, const float* __restrict__ tsw,
                             const float* __restrict__ tsb, float* __restrict__ sOut, int fin) {
  int idx = blockIdx.x * blockDim.x + threadIdx.x;
  if (idx >= NB * fin) return;
  int b = idx / fin, i = idx - b * fin;
  const float* wr = wlat + b * WDIM;
  const float* tw = tsw + (size_t)i * WDIM;
  float acc = 0.0f;
  for (int j = 0; j < WDIM; ++j) acc += wr[j] * tw[j];
  sOut[idx] = acc * 0.0625f + tsb[i];     // 1/sqrt(256) exactly
}

// ---------------- demod: sig[b,o] = rsqrt( sum_{c,t} (cw*cscale*s[b,c])^2 + eps ) -------------
__global__ void sigma_kernel(const float* __restrict__ cw, const float* __restrict__ s,
                             float* __restrict__ sig, int O, int C, float cscale) {
  int idx = blockIdx.x * blockDim.x + threadIdx.x;
  if (idx >= NB * O) return;
  int b = idx / O, o = idx - b * O;
  const float* wr = cw + (size_t)o * C * 9;
  const float* sb = s + b * C;
  float sum = 0.0f;
  for (int c = 0; c < C; ++c) {
    float sv = sb[c] * cscale;
    #pragma unroll
    for (int t = 0; t < 9; ++t) { float v = wr[c * 9 + t] * sv; sum += v * v; }
  }
  sig[idx] = rsqrtf(sum + 1e-8f);
}

// ---------------- pack modulated weights into A-fragment order, f16 -------------------------
// Layout: [b, oTile, tap, cChunk, lane(32), elem(16)]  -> conv reads one v16h per lane/chunk.
__global__ void pack_wmod_kernel(const float* __restrict__ cw, const float* __restrict__ s,
                                 const float* __restrict__ sig, _Float16* __restrict__ ap,
                                 int O, int C, float cscale) {
  const int nCC = C >> 5, OT = O >> 4;
  const int total = NB * OT * 9 * nCC * 512;
  int g = blockIdx.x * blockDim.x + threadIdx.x;
  if (g >= total) return;
  const int e    = g & 15;
  const int lane = (g >> 4) & 31;
  int chunk = g >> 9;
  const int cc = chunk % nCC; chunk /= nCC;
  const int t  = chunk % 9;   chunk /= 9;
  const int ot = chunk % OT;
  const int b  = chunk / OT;
  const int half = lane >> 4, mn = lane & 15;
  const int o = (ot << 4) + mn;
  const int j = e >> 1;
  // A fragment K index: contiguous order k = 16*half + e (matches B read)
  const int ka = (half << 4) + e + (j & 0);
  const int c = (cc << 5) + ka;
  const float v = cw[((size_t)o * C + c) * 9 + t] * s[b * C + c] * cscale * sig[b * O + o];
  ap[g] = (_Float16)v;
}

// ---------------- modulated 3x3 conv: 9 taps x channel-GEMM on v_wmma_f32_16x16x32_f16 ------
// x channels-last f16 [B,HW,C]; packed A from pack_wmod; grid = (HW/16, O/16, B), block = 32.
__global__ __launch_bounds__(32) void modconv3x3_wmma(
    const _Float16* __restrict__ xh,   // [B,HW,C] f16
    const _Float16* __restrict__ ap,   // packed wmod
    const float* __restrict__ noise,   // [B,1,H,W]
    const float* __restrict__ snp,     // [1]
    const float* __restrict__ bias,    // [O]
    _Float16* __restrict__ yh,         // [B,HW,O] f16
    int C, int O, int H, int W)
{
  const int lane = threadIdx.x;
  const int half = lane >> 4, mn = lane & 15;
  const int b  = blockIdx.z;
  const int HW = H * W;
  const int p  = (blockIdx.x << 4) + mn;       // B-matrix column (pixel)
  const int ph = p / W, pw = p - ph * W;
  const int nCC = C >> 5;

  const _Float16* apl = ap + ((((size_t)b * gridDim.y + blockIdx.y) * 9 * nCC) << 9) + (lane << 4);
  const _Float16* xb  = xh + (size_t)b * HW * C + (half << 4);

  v8f acc = {};
  #pragma unroll
  for (int t = 0; t < 9; ++t) {
    const int dy = t / 3 - 1, dx = t % 3 - 1;      // compile-time (t unrolled)
    const int ih = ph + dy, iw = pw + dx;
    const bool ok = ((unsigned)ih < (unsigned)H) && ((unsigned)iw < (unsigned)W);
    const _Float16 mh = ok ? (_Float16)1.0f : (_Float16)0.0f;   // branchless zero-pad
    const _Float16* xp = xb + (size_t)(ok ? (ih * W + iw) : 0) * C;
    for (int c0 = 0; c0 < C; c0 += 32) {
      v16h Af = *(const v16h*)apl;  apl += 512;
      v16h Bf = *(const v16h*)(xp + c0);
      #pragma unroll
      for (int e2 = 0; e2 < 16; ++e2) Bf[e2] = Bf[e2] * mh;
      acc = __builtin_amdgcn_wmma_f32_16x16x32_f16(false, Af, false, Bf,
                                                   (short)0, acc, false, false);
    }
  }

  const float nv = snp[0] * noise[(size_t)b * HW + p];
  const int obase = (blockIdx.y << 4) + (half << 3);   // D: lanes0-15 M=r, lanes16-31 M=r+8
  const v8f bb = *(const v8f*)(bias + obase);
  v8h outv;
  #pragma unroll
  for (int r = 0; r < 8; ++r)
    outv[r] = (_Float16)leaky_f(acc[r] + nv + bb[r]);
  *(v8h*)(yh + ((size_t)b * HW + p) * O + obase) = outv;
}

// ---------------- to_rgb: 1x1 conv (O=1, no demod), leaky, optional accumulate ---------------
__global__ void rgb_kernel(const _Float16* __restrict__ xh, const float* __restrict__ s,
                           const float* __restrict__ cw, const float* __restrict__ bias,
                           float* __restrict__ rgb, int C, int HW, float cscale, int accumulate) {
  int idx = blockIdx.x * blockDim.x + threadIdx.x;
  if (idx >= NB * HW) return;
  int b = idx / HW, p = idx - b * HW;
  const _Float16* xp = xh + ((size_t)b * HW + p) * C;
  const float* sp = s + b * C;
  float acc = 0.0f;
  for (int c = 0; c < C; ++c) acc += cw[c] * sp[c] * (float)xp[c];
  float v = leaky_f(acc * cscale + bias[0]);
  rgb[idx] = (accumulate ? rgb[idx] : 0.0f) + v;
}

// ------------- bilinear 2x upsample, channels-last f16, half-pixel centers -------------------
__global__ void up2cl_kernel(const _Float16* __restrict__ in, _Float16* __restrict__ out,
                             int C, int H, int W) {
  const int OW = 2 * W, OH = 2 * H;
  long long total = (long long)NB * OH * OW * C;
  long long idx = (long long)blockIdx.x * blockDim.x + threadIdx.x;
  if (idx >= total) return;
  int c = (int)(idx % C);
  long long t = idx / C;
  int ox = (int)(t % OW); t /= OW;
  int oy = (int)(t % OH);
  int b  = (int)(t / OH);
  float fy = (oy + 0.5f) * 0.5f - 0.5f;
  float fx = (ox + 0.5f) * 0.5f - 0.5f;
  int y0 = (int)floorf(fy); float wy = fy - (float)y0;
  int x0 = (int)floorf(fx); float wx = fx - (float)x0;
  int y0c = min(max(y0, 0), H - 1), y1c = min(max(y0 + 1, 0), H - 1);
  int x0c = min(max(x0, 0), W - 1), x1c = min(max(x0 + 1, 0), W - 1);
  const _Float16* pin = in + (size_t)b * H * W * C;
  float v = (1.0f - wy) * ((1.0f - wx) * (float)pin[(size_t)(y0c * W + x0c) * C + c] +
                           wx          * (float)pin[(size_t)(y0c * W + x1c) * C + c]) +
            wy          * ((1.0f - wx) * (float)pin[(size_t)(y1c * W + x0c) * C + c] +
                           wx          * (float)pin[(size_t)(y1c * W + x1c) * C + c]);
  out[idx] = (_Float16)v;
}

// ---------------- bilinear 2x upsample, f32 (rgb, 1 channel) ---------------------------------
__global__ void up2_kernel(const float* __restrict__ in, float* __restrict__ out,
                           int BC, int H, int W) {
  int OW = 2 * W, OH = 2 * H;
  long long total = (long long)BC * OH * OW;
  long long idx = (long long)blockIdx.x * blockDim.x + threadIdx.x;
  if (idx >= total) return;
  int ox = (int)(idx % OW);
  long long t = idx / OW;
  int oy = (int)(t % OH);
  int bc = (int)(t / OH);
  float fy = (oy + 0.5f) * 0.5f - 0.5f;
  float fx = (ox + 0.5f) * 0.5f - 0.5f;
  int y0 = (int)floorf(fy); float wy = fy - (float)y0;
  int x0 = (int)floorf(fx); float wx = fx - (float)x0;
  int y0c = min(max(y0, 0), H - 1), y1c = min(max(y0 + 1, 0), H - 1);
  int x0c = min(max(x0, 0), W - 1), x1c = min(max(x0 + 1, 0), W - 1);
  const float* pin = in + (size_t)bc * H * W;
  float v = (1.0f - wy) * ((1.0f - wx) * pin[y0c * W + x0c] + wx * pin[y0c * W + x1c]) +
            wy          * ((1.0f - wx) * pin[y1c * W + x0c] + wx * pin[y1c * W + x1c]);
  out[idx] = v;
}

// ---------------- broadcast const across batch, NCHW f32 -> channels-last f16 ----------------
__global__ void bcast_cl_kernel(const float* __restrict__ cst, _Float16* __restrict__ xh) {
  int idx = blockIdx.x * blockDim.x + threadIdx.x;  // NB*256(pix)*256(ch)
  if (idx >= NB * 256 * 256) return;
  int c = idx & 255;
  int p = (idx >> 8) & 255;
  xh[idx] = (_Float16)cst[c * 256 + p];
}

// ---------------- final tanh ----------------
__global__ void tanh_kernel(const float* __restrict__ in, float* __restrict__ out, int n) {
  int idx = blockIdx.x * blockDim.x + threadIdx.x;
  if (idx >= n) return;
  out[idx] = tanhf(in[idx]);
}

// ===================================================================================
extern "C" void kernel_launch(void* const* d_in, const int* in_sizes, int n_in,
                              void* d_out, int out_size, void* d_ws, size_t ws_size,
                              hipStream_t stream) {
  (void)in_sizes; (void)n_in; (void)out_size; (void)ws_size;

  // ---- input mapping (jax pytree / sorted-dict flattening of `params`) ----
  // 0:w 1:noise0 2:n1a 3:n1b 4:n2a 5:n2b 6:n3a 7:n3b 8:n4a 9:n4b
  // blocks[k] base = 10+14k: rgb{b,cw,ts_b,ts_w}=+0..+3  sb1{b,cw,sn,ts_b,ts_w}=+4..+8
  //                          sb2{b,cw,sn,ts_b,ts_w}=+9..+13
  // 66:const  67..70:rgb0{b,cw,ts_b,ts_w}  71..75:sb0{b,cw,sn,ts_b,ts_w}
  auto F = [&](int i) { return (const float*)d_in[i]; };
  const float* W5 = F(0);

  // ---- workspace layout ----
  const size_t X_ELEMS  = (size_t)NB * 64 * 256 * 256;   // 67,108,864 halfs (largest up2 output)
  const size_t AP_ELEMS = (size_t)NB * 256 * 256 * 9;    // 9,437,184 halfs (largest wmod)
  const size_t R_ELEMS  = (size_t)NB * 256 * 256;        // 1,048,576 floats
  _Float16* XH0 = (_Float16*)d_ws;
  _Float16* XH1 = XH0 + X_ELEMS;
  _Float16* AP  = XH1 + X_ELEMS;
  float* R0  = (float*)(AP + AP_ELEMS);
  float* R1  = R0 + R_ELEMS;
  float* S   = R1 + R_ELEMS;                             // [16, <=256]
  float* SIG = S + NB * 256;                             // [16, <=256]

  auto cdiv = [](long long a, long long b) { return (unsigned)((a + b - 1) / b); };
  auto csc3 = [](int C) { return 1.0f / sqrtf((float)C * 9.0f); };
  auto csc1 = [](int C) { return 1.0f / sqrtf((float)C); };

  auto style = [&](const float* wl, const float* tsw, const float* tsb, int fin) {
    style_kernel<<<cdiv(NB * fin, 256), 256, 0, stream>>>(wl, tsw, tsb, S, fin);
  };
  auto sigma = [&](const float* cw, int O, int C) {
    sigma_kernel<<<cdiv(NB * O, 128), 128, 0, stream>>>(cw, S, SIG, O, C, csc3(C));
  };
  auto pack = [&](const float* cw, int O, int C) {
    long long n = (long long)NB * (O >> 4) * 9 * (C >> 5) * 512;
    pack_wmod_kernel<<<cdiv(n, 256), 256, 0, stream>>>(cw, S, SIG, AP, O, C, csc3(C));
  };
  auto modconv = [&](const _Float16* x, const float* noi, const float* sn,
                     const float* bias, _Float16* y, int C, int O, int H) {
    dim3 grid(H * H / 16, O / 16, NB);
    modconv3x3_wmma<<<grid, 32, 0, stream>>>(x, AP, noi, sn, bias, y, C, O, H, H);
  };
  auto torgb = [&](const _Float16* x, const float* cw, const float* bias, float* r,
                   int C, int H, int acc) {
    rgb_kernel<<<cdiv((long long)NB * H * H, 256), 256, 0, stream>>>(x, S, cw, bias, r, C, H * H, csc1(C), acc);
  };
  auto up2x = [&](const _Float16* in, _Float16* out, int C, int H) {
    up2cl_kernel<<<cdiv((long long)NB * C * 4 * H * H, 256), 256, 0, stream>>>(in, out, C, H, H);
  };
  auto up2r = [&](const float* in, float* out, int H) {
    up2_kernel<<<cdiv((long long)NB * 4 * H * H, 256), 256, 0, stream>>>(in, out, NB, H, H);
  };

  const int feat[5] = {256, 256, 128, 64, 32};

  // ---- stage 0 (res 16, C=256) ----
  bcast_cl_kernel<<<cdiv((long long)NB * 256 * 256, 256), 256, 0, stream>>>(F(66), XH0);
  style(W5 + 0 * NB * WDIM, F(75), F(74), 256);
  sigma(F(72), 256, 256);
  pack(F(72), 256, 256);
  modconv(XH0, F(1), F(73), F(71), XH1, 256, 256, 16);
  style(W5 + 0 * NB * WDIM, F(70), F(69), 256);
  torgb(XH1, F(68), F(67), R0, 256, 16, /*acc=*/0);

  _Float16* xcur = XH1; _Float16* xalt = XH0;
  float* rcur = R0; float* ralt = R1;
  int H = 16, Cin = 256;

  // ---- blocks 1..4 ----
  for (int k = 0; k < 4; ++k) {
    const int base = 10 + 14 * k;
    const int Cout = feat[k + 1];
    const float* wl = W5 + (size_t)(k + 1) * NB * WDIM;
    const float* na = F(2 + 2 * k);
    const float* nb = F(3 + 2 * k);
    const int H2 = H * 2;

    up2x(xcur, xalt, Cin, H);                           { _Float16* t = xcur; xcur = xalt; xalt = t; }

    style(wl, F(base + 8), F(base + 7), Cin);
    sigma(F(base + 5), Cout, Cin);
    pack(F(base + 5), Cout, Cin);
    modconv(xcur, na, F(base + 6), F(base + 4), xalt, Cin, Cout, H2);
    { _Float16* t = xcur; xcur = xalt; xalt = t; }

    style(wl, F(base + 13), F(base + 12), Cout);
    sigma(F(base + 10), Cout, Cout);
    pack(F(base + 10), Cout, Cout);
    modconv(xcur, nb, F(base + 11), F(base + 9), xalt, Cout, Cout, H2);
    { _Float16* t = xcur; xcur = xalt; xalt = t; }

    up2r(rcur, ralt, H);                                { float* t = rcur; rcur = ralt; ralt = t; }
    style(wl, F(base + 3), F(base + 2), Cout);
    torgb(xcur, F(base + 1), F(base + 0), rcur, Cout, H2, /*acc=*/1);

    H = H2; Cin = Cout;
  }

  tanh_kernel<<<cdiv(NB * 256 * 256, 256), 256, 0, stream>>>(rcur, (float*)d_out, NB * 256 * 256);
}
